// MultiHeadAttention_47090021433500
// MI455X (gfx1250) — matrix operand order repaired and checked by reading, not compiled, not run on hardware
//
#include <hip/hip_runtime.h>
#include <hip/hip_bf16.h>

typedef __attribute__((ext_vector_type(16))) __bf16 v16bf;
typedef __attribute__((ext_vector_type(8)))  float  v8f;
typedef int v4i __attribute__((ext_vector_type(4)));
typedef v4i __attribute__((address_space(1))) *g_v4i_p;   // global int4*
typedef v4i __attribute__((address_space(3))) *l_v4i_p;   // LDS int4*

#define D_MODEL 1024
#define HEADS   16
#define DK      64
#define SEQ     2048
#define BATCH   2

#if defined(__has_builtin)
#if __has_builtin(__builtin_amdgcn_global_load_async_to_lds_b128) && \
    __has_builtin(__builtin_amdgcn_s_wait_asynccnt)
#define HAS_ASYNC 1
#endif
#endif
#ifndef HAS_ASYNC
#define HAS_ASYNC 0
#endif

__device__ __forceinline__ unsigned short f2bf(float f) {
    union { float f; unsigned int u; } x; x.f = f;
    unsigned int u = x.u;
    unsigned int r = u + 0x7FFFu + ((u >> 16) & 1u);   // round-to-nearest-even
    return (unsigned short)(r >> 16);
}

// 16-byte async (or sync fallback) copy: global bf16 -> LDS bf16
__device__ __forceinline__ void copy16_g2l(unsigned short* dst, const unsigned short* src) {
#if HAS_ASYNC
    __builtin_amdgcn_global_load_async_to_lds_b128((g_v4i_p)src, (l_v4i_p)dst, 0, 0);
#else
    *(uint4*)dst = *(const uint4*)src;
#endif
}

__device__ __forceinline__ void async_wait_all() {
#if HAS_ASYNC
    __builtin_amdgcn_s_wait_asynccnt(0);
#endif
}

// ---------------------------------------------------------------------------
// One-shot: transpose + convert the four 1024x1024 f32 weights to bf16,
// N-major: Wt[z][n][k] = bf16(W_z[k][n]).  grid(32,32,4), 256 threads.
// ---------------------------------------------------------------------------
__global__ __launch_bounds__(256) void transpose_w_bf16(
    const float* __restrict__ W0, const float* __restrict__ W1,
    const float* __restrict__ W2, const float* __restrict__ W3,
    unsigned short* __restrict__ Wt)
{
    __shared__ float tile[32][33];
    const float* W = (blockIdx.z == 0) ? W0 : (blockIdx.z == 1) ? W1
                   : (blockIdx.z == 2) ? W2 : W3;
    unsigned short* T = Wt + (size_t)blockIdx.z * D_MODEL * D_MODEL;
    const int k0 = blockIdx.y * 32, n0 = blockIdx.x * 32;
    const int tx = threadIdx.x & 31, ty = threadIdx.x >> 5;
#pragma unroll
    for (int i = 0; i < 4; ++i) {
        int k = ty + i * 8;
        tile[k][tx] = W[(size_t)(k0 + k) * D_MODEL + n0 + tx];
    }
    __syncthreads();
#pragma unroll
    for (int i = 0; i < 4; ++i) {
        int n = ty + i * 8;
        T[(size_t)(n0 + n) * D_MODEL + k0 + tx] = f2bf(tile[tx][n]);
    }
}

// ---------------------------------------------------------------------------
// Tiled GEMM: C = A(Mx1024) * W(1024x1024) + bias via v_wmma_f32_16x16x32_bf16.
// W is pre-converted bf16 N-major (Wt[n][k]).  Block tile 128x64, BK=32,
// 256 threads = 8 waves, each wave -> 16x64 of C.  Double-buffered LDS with
// async global->LDS staging for all-bf16 tiles.
// A_IS_BF16: 0 = A is f32 (convert while staging), 1 = A is bf16 (async).
// EPI: 0 = f32 row-major out (+bias)
//      1 = bf16 head-split out  [b,h,s,dk]   (Q / K)
//      2 = bf16 head-split outT [b,h,dk,s]   (V transposed)
// ---------------------------------------------------------------------------
template <int A_IS_BF16, int EPI>
__global__ __launch_bounds__(256) void gemm_bf16_wmma(
    const void* __restrict__ Av, const unsigned short* __restrict__ Wt,
    const float* __restrict__ bias, void* __restrict__ Out)
{
    __shared__ __align__(16) unsigned short As[2][128 * 32];   // [m][k]
    __shared__ __align__(16) unsigned short Bt[2][64 * 40];    // [n][k], padded

    const int t     = threadIdx.x;
    const int mBase = blockIdx.y * 128;
    const int nBase = blockIdx.x * 64;
    const int w     = t >> 5;
    const int lane  = t & 31;
    const int ln    = lane & 15;
    const int half  = lane >> 4;

    v8f acc[4];
#pragma unroll
    for (int i = 0; i < 4; ++i) acc[i] = (v8f){0.f,0.f,0.f,0.f,0.f,0.f,0.f,0.f};

    auto stage = [&](int buf, int k0) {
        // ---- B tile: 64 rows x 32 bf16, contiguous rows of Wt -> async ----
        {
            int n = t >> 2, ch = t & 3;                     // 256 x 16B = 4KB
            copy16_g2l(&Bt[buf][n * 40 + ch * 8],
                       Wt + (size_t)(nBase + n) * D_MODEL + k0 + ch * 8);
        }
        // ---- A tile: 128 rows x 32 elems ----
        if (A_IS_BF16) {
            const unsigned short* A16 = (const unsigned short*)Av;
#pragma unroll
            for (int i = 0; i < 2; ++i) {
                int id = t + i * 256, r = id >> 2, ch = id & 3;   // 512 x 16B
                copy16_g2l(&As[buf][r * 32 + ch * 8],
                           A16 + (size_t)(mBase + r) * D_MODEL + k0 + ch * 8);
            }
        } else {
            const float* A32 = (const float*)Av;
#pragma unroll
            for (int i = 0; i < 4; ++i) {
                int vv = t + i * 256;                   // 0..1023 (float4)
                int r  = vv >> 3;
                int c4 = (vv & 7) << 2;
                float4 f = *(const float4*)(A32 + (size_t)(mBase + r) * D_MODEL + k0 + c4);
                unsigned short* p = &As[buf][r * 32 + c4];
                p[0] = f2bf(f.x); p[1] = f2bf(f.y); p[2] = f2bf(f.z); p[3] = f2bf(f.w);
            }
        }
    };

    stage(0, 0);
    int buf = 0;
    for (int k0 = 0; k0 < D_MODEL; k0 += 32) {
        async_wait_all();               // this wave's async tile writes done
        __syncthreads();                // all waves' tile writes visible
        if (k0 + 32 < D_MODEL) stage(buf ^ 1, k0 + 32);   // overlap next tile

        // ---- A fragment: 16x32, lane row = w*16+ln, K = {half*8.., 16+half*8..}
        const unsigned short* arow = &As[buf][(w * 16 + ln) * 32];
        v16bf af;
        ((uint4*)&af)[0] = *(const uint4*)(arow + half * 8);
        ((uint4*)&af)[1] = *(const uint4*)(arow + 16 + half * 8);
#pragma unroll
        for (int nt = 0; nt < 4; ++nt) {
            // ---- B fragment: 32x16, lane col = nt*16+ln, K = {half*8.., 16+half*8..}
            const unsigned short* brow = &Bt[buf][(nt * 16 + ln) * 40 + half * 8];
            v16bf bf_;
            ((uint4*)&bf_)[0] = *(const uint4*)(brow);
            ((uint4*)&bf_)[1] = *(const uint4*)(brow + 16);
            acc[nt] = __builtin_amdgcn_wmma_f32_16x16x32_bf16(
                false, af, false, bf_, (short)0, acc[nt], false, false);
        }
        buf ^= 1;
    }

    // ---- epilogue: C layout -> VGPR i = row (i + 8*half), lane%16 = col ----
#pragma unroll
    for (int nt = 0; nt < 4; ++nt) {
        int   col = nBase + nt * 16 + ln;
        float bv  = bias[col];
#pragma unroll
        for (int i = 0; i < 8; ++i) {
            int   gm  = mBase + w * 16 + half * 8 + i;
            float val = acc[nt][i] + bv;
            if (EPI == 0) {
                ((float*)Out)[(size_t)gm * D_MODEL + col] = val;
            } else {
                int bb = gm >> 11, s = gm & (SEQ - 1);
                int h  = col >> 6, dk = col & 63;
                if (EPI == 1)
                    ((unsigned short*)Out)[(((size_t)(bb * HEADS + h) * SEQ + s) << 6) + dk] = f2bf(val);
                else
                    ((unsigned short*)Out)[((size_t)(bb * HEADS + h) * DK + dk) * SEQ + s] = f2bf(val);
            }
        }
    }
}

// ---------------------------------------------------------------------------
// Flash attention: one wave per 16-query tile; key blocks of 32; online
// softmax; scores and PV both via v_wmma_f32_16x16x32_bf16.
// Qh,Kh: [b,h,s,dk] bf16;  VhT: [b,h,dk,s] bf16;  Ctx out: [b,s,d_model] bf16
// ---------------------------------------------------------------------------
__global__ __launch_bounds__(128) void flash_attn_wmma(
    const unsigned short* __restrict__ Qh, const unsigned short* __restrict__ Kh,
    const unsigned short* __restrict__ VhT, const int* __restrict__ mask,
    unsigned short* __restrict__ Ctx)
{
    __shared__ __align__(16) unsigned short Plds[4 * 16 * 32];   // 1KB per wave

    const int t    = threadIdx.x;
    const int w    = t >> 5;
    const int lane = t & 31;
    const int ln   = lane & 15;
    const int half = lane >> 4;

    const int g  = blockIdx.x * 4 + w;     // global q-tile id
    const int bh = g >> 7;                 // 128 q-tiles per (b,h)
    const int qt = g & 127;
    const int b  = bh >> 4;
    const int h  = bh & 15;
    const int s0 = qt << 4;

    const unsigned short* Qbase = Qh  + (size_t)bh * SEQ * DK;
    const unsigned short* Kbase = Kh  + (size_t)bh * SEQ * DK;
    const unsigned short* Vbase = VhT + (size_t)bh * DK * SEQ;
    const int*            mrow  = mask + b * SEQ;

    // Q A-fragments for both dk k-steps (reused across all key blocks)
    v16bf qf[2];
    {
        const unsigned short* qrow = Qbase + (size_t)(s0 + ln) * DK;
#pragma unroll
        for (int ks = 0; ks < 2; ++ks) {
            ((uint4*)&qf[ks])[0] = *(const uint4*)(qrow + ks * 32 + half * 8);
            ((uint4*)&qf[ks])[1] = *(const uint4*)(qrow + ks * 32 + 16 + half * 8);
        }
    }

    v8f o[4];
#pragma unroll
    for (int i = 0; i < 4; ++i) o[i] = (v8f){0.f,0.f,0.f,0.f,0.f,0.f,0.f,0.f};
    float mrun[8], lrun[8];
#pragma unroll
    for (int i = 0; i < 8; ++i) { mrun[i] = -3.0e38f; lrun[i] = 0.f; }

    unsigned short* P = &Plds[w * 16 * 32];

    for (int j0 = 0; j0 < SEQ; j0 += 32) {
        // prefetch next key block's K rows / V rows (global_prefetch_b8)
        if (j0 + 32 < SEQ) {
            __builtin_prefetch(Kbase + (size_t)(j0 + 32 + lane) * DK, 0, 1);
            __builtin_prefetch(Vbase + (size_t)lane * SEQ + j0 + 32, 0, 1);
            __builtin_prefetch(Vbase + (size_t)(32 + lane) * SEQ + j0 + 32, 0, 1);
        }

        // ---- scores S = Q * K^T for 16x32 block ----
        v8f sc[2];
#pragma unroll
        for (int nt = 0; nt < 2; ++nt) {
            sc[nt] = (v8f){0.f,0.f,0.f,0.f,0.f,0.f,0.f,0.f};
            const unsigned short* krow = Kbase + (size_t)(j0 + nt * 16 + ln) * DK;
#pragma unroll
            for (int ks = 0; ks < 2; ++ks) {
                v16bf kf;
                ((uint4*)&kf)[0] = *(const uint4*)(krow + ks * 32 + half * 8);
                ((uint4*)&kf)[1] = *(const uint4*)(krow + ks * 32 + 16 + half * 8);
                sc[nt] = __builtin_amdgcn_wmma_f32_16x16x32_bf16(
                    false, qf[ks], false, kf, (short)0, sc[nt], false, false);
            }
        }

        // ---- scale + mask + online softmax (row reduce over 16-lane halves) --
        const int mk0 = mrow[j0 + ln];
        const int mk1 = mrow[j0 + 16 + ln];
        float pblk[2][8];
#pragma unroll
        for (int i = 0; i < 8; ++i) {
            float a0 = mk0 ? sc[0][i] * 0.125f : -1.0e9f;
            float a1 = mk1 ? sc[1][i] * 0.125f : -1.0e9f;
            float mx = fmaxf(a0, a1);
#pragma unroll
            for (int d = 8; d >= 1; d >>= 1) mx = fmaxf(mx, __shfl_xor(mx, d, 32));
            float mnew = fmaxf(mrun[i], mx);
            float p0 = __expf(a0 - mnew);
            float p1 = __expf(a1 - mnew);
            float ps = p0 + p1;
#pragma unroll
            for (int d = 8; d >= 1; d >>= 1) ps += __shfl_xor(ps, d, 32);
            float f = __expf(mrun[i] - mnew);
            lrun[i] = lrun[i] * f + ps;
            mrun[i] = mnew;
#pragma unroll
            for (int nt = 0; nt < 4; ++nt) o[nt][i] *= f;
            pblk[0][i] = p0; pblk[1][i] = p1;
        }

        // ---- re-layout P (C layout -> A fragment) via private LDS slab ----
#pragma unroll
        for (int nt = 0; nt < 2; ++nt)
#pragma unroll
            for (int i = 0; i < 8; ++i)
                P[(half * 8 + i) * 32 + nt * 16 + ln] = f2bf(pblk[nt][i]);

        v16bf pf;
        {
            const unsigned short* prow = &P[ln * 32];
            ((uint4*)&pf)[0] = *(const uint4*)(prow + half * 8);
            ((uint4*)&pf)[1] = *(const uint4*)(prow + 16 + half * 8);
        }

        // ---- O += P * V  (V^T rows are s-contiguous -> direct B fragments) --
#pragma unroll
        for (int nt = 0; nt < 4; ++nt) {
            const unsigned short* vrow = Vbase + (size_t)(nt * 16 + ln) * SEQ + j0 + half * 8;
            v16bf vf;
            ((uint4*)&vf)[0] = *(const uint4*)(vrow);
            ((uint4*)&vf)[1] = *(const uint4*)(vrow + 16);
            o[nt] = __builtin_amdgcn_wmma_f32_16x16x32_bf16(
                false, pf, false, vf, (short)0, o[nt], false, false);
        }
    }

    // ---- normalize and write ctx in [b,s,d_model] bf16 ----
#pragma unroll
    for (int i = 0; i < 8; ++i) lrun[i] = 1.0f / lrun[i];
#pragma unroll
    for (int nt = 0; nt < 4; ++nt)
#pragma unroll
        for (int i = 0; i < 8; ++i) {
            int s = s0 + half * 8 + i;
            Ctx[(size_t)(b * SEQ + s) * D_MODEL + h * DK + nt * 16 + ln] =
                f2bf(o[nt][i] * lrun[i]);
        }
}

// ---------------------------------------------------------------------------
extern "C" void kernel_launch(void* const* d_in, const int* in_sizes, int n_in,
                              void* d_out, int out_size, void* d_ws, size_t ws_size,
                              hipStream_t stream)
{
    const float* q  = (const float*)d_in[0];
    const float* k  = (const float*)d_in[1];
    const float* v  = (const float*)d_in[2];
    const int*   sm = (const int*)  d_in[3];
    const float* Wq = (const float*)d_in[4];
    const float* bq = (const float*)d_in[5];
    const float* Wk = (const float*)d_in[6];
    const float* bk = (const float*)d_in[7];
    const float* Wv = (const float*)d_in[8];
    const float* bv = (const float*)d_in[9];
    const float* Wo = (const float*)d_in[10];
    const float* bo = (const float*)d_in[11];
    float*       out = (float*)d_out;

    const int M = BATCH * SEQ;                         // 4096
    unsigned short* Qh  = (unsigned short*)d_ws;       // 8 MB each
    unsigned short* Kh  = Qh  + (size_t)M * D_MODEL;
    unsigned short* VhT = Kh  + (size_t)M * D_MODEL;
    unsigned short* Ctx = VhT + (size_t)M * D_MODEL;
    unsigned short* Wt  = Ctx + (size_t)M * D_MODEL;   // 4x 2MB bf16 N-major
    unsigned short* Wtq = Wt;
    unsigned short* Wtk = Wt + (size_t)1 * D_MODEL * D_MODEL;
    unsigned short* Wtv = Wt + (size_t)2 * D_MODEL * D_MODEL;
    unsigned short* Wto = Wt + (size_t)3 * D_MODEL * D_MODEL;

    transpose_w_bf16<<<dim3(32, 32, 4), 256, 0, stream>>>(Wq, Wk, Wv, Wo, Wt);

    dim3 ggrid(D_MODEL / 64, M / 128);                 // 16 x 32 blocks
    gemm_bf16_wmma<0, 1><<<ggrid, 256, 0, stream>>>(q, Wtq, bq, Qh);
    gemm_bf16_wmma<0, 1><<<ggrid, 256, 0, stream>>>(k, Wtk, bk, Kh);
    gemm_bf16_wmma<0, 2><<<ggrid, 256, 0, stream>>>(v, Wtv, bv, VhT);

    flash_attn_wmma<<<dim3((BATCH * HEADS * (SEQ / 16)) / 4), 128, 0, stream>>>(
        Qh, Kh, VhT, sm, Ctx);

    gemm_bf16_wmma<1, 0><<<ggrid, 256, 0, stream>>>(Ctx, Wto, bo, out);
}
